// PPGDPOPolicy_40862318854415
// MI455X (gfx1250) — matrix operand order repaired and checked by reading, not compiled, not run on hardware
//
#include <hip/hip_runtime.h>
#include <hip/hip_bf16.h>
#include <math.h>

typedef __bf16 bf16_t;
typedef __attribute__((ext_vector_type(16))) __bf16 v16bf;
typedef __attribute__((ext_vector_type(8)))  float  v8f;

#define SPW    64      // samples (paths) per workgroup
#define NWG    64      // 4096 / 64
#define TPB    128     // 4 wave32 per workgroup
#define MSTEPS 60

// model constants
#define T_F      1.5f
#define DT_F     0.025f
#define HALF_F   0.0125f
#define SDT_F    0.15811388f     // sqrt(DT)
#define SHALF_F  0.11180340f     // sqrt(DT/2)
#define RHO_F    0.3f
#define ACORR_F  0.95393920f     // sqrt(1-rho^2)
#define INVS2_F  0.70710678f
#define SIGMA_F  0.2f
#define SIGALP_F 0.16f           // SIGMA*ALPHA
#define SIG2_F   0.04f
#define KAPPA_F  1.2f
#define SIGY_F   0.3f
#define RRATE_F  0.02f
#define LOGLB_F  -6.90775528f    // log(0.001)

struct SharedBlk {
  bf16_t h1j[5][SPW][128];   // layer-1 jet activations (bf16 for WMMA A)  80 KB
  bf16_t w2T[128][128];      // w2 transposed [n][k] (WMMA B)              32 KB
  float  w1s[3][128];
  float  b1s[128], b2s[128], w3s[128];
  float  xj[5][SPW];         // W-input jets (value, a, b, aa, ab)
  float  xy[SPW];            // Y-input value
  float  piJ[5][SPW];        // policy jets out
};

union Frag { v16bf v; unsigned int u[8]; };

// Branchless tanh: h = 1 - 2*rcp(exp(2x)+1). v_exp_f32 and v_rcp_f32 are
// TRANS ops that co-execute with the WMMA pipe; no EXEC branches, no IEEE
// division expansion (v_div_scale/div_fmas).
__device__ __forceinline__ float fast_tanh(float x) {
  float e = __expf(2.f * x);
  return 1.f - 2.f * __builtin_amdgcn_rcpf(e + 1.f);
}

// A fragment: 16x32 bf16, row m = lane&15; K split per ISA 7.12.2 table.
__device__ __forceinline__ v16bf load_a_frag(const bf16_t* base, int lane, int kc) {
  Frag f;
  int m  = lane & 15;
  int kg = (lane < 16) ? 0 : 8;
  const bf16_t* p = base + m * 128 + kc * 32;
#pragma unroll
  for (int v = 0; v < 8; ++v) {
    int k = (v < 4) ? (kg + 2 * v) : (16 + kg + 2 * (v - 4));
    f.u[v] = *(const unsigned int*)(p + k);
  }
  return f.v;
}

// B fragment: 32x16 bf16, col n = lane&15 (+tile); K split per ISA 7.12.2 table (same as A).
__device__ __forceinline__ v16bf load_b_frag(const bf16_t* w2T, int lane, int nt, int kc) {
  Frag f;
  int n    = (lane & 15) + nt * 16;
  int koff = (lane < 16) ? 0 : 8;
  const bf16_t* p = w2T + n * 128 + kc * 32 + koff;
#pragma unroll
  for (int v = 0; v < 8; ++v) f.u[v] = *(const unsigned int*)(p + ((v < 4) ? 2 * v : 16 + 2 * (v - 4)));
  return f.v;
}

// Full-block 2nd-order-jet MLP eval. xj/xy -> piJ. tmt, cYin uniform.
__device__ __forceinline__ void mlp_eval(SharedBlk& sb, int tid, float tmt,
                                         float cYin, float b3v) {
  __syncthreads();   // xj/xy visible
  // ---- layer 1 (K=3) in VALU, tanh-jet, write bf16 jets ----
  {
    int s = tid >> 1, hb = (tid & 1) << 6;
    float x0 = sb.xj[0][s], xa = sb.xj[1][s], xb = sb.xj[2][s];
    float xaa = sb.xj[3][s], xab = sb.xj[4][s];
    float yv = sb.xy[s];
#pragma unroll 4
    for (int i = 0; i < 64; ++i) {
      int h = hb + i;
      float w10 = sb.w1s[0][h], w11 = sb.w1s[1][h], w12 = sb.w1s[2][h];
      float z   = x0 * w10 + tmt * w11 + yv * w12 + sb.b1s[h];
      float za  = xa * w10;
      float zb  = xb * w10 + cYin * w12;
      float zaa = xaa * w10;
      float zab = xab * w10;
      float hv = fast_tanh(z);
      float s1 = 1.f - hv * hv;
      float t2 = -2.f * hv * s1;
      sb.h1j[0][s][h] = (bf16_t)hv;
      sb.h1j[1][s][h] = (bf16_t)(s1 * za);
      sb.h1j[2][s][h] = (bf16_t)(s1 * zb);
      sb.h1j[3][s][h] = (bf16_t)(s1 * zaa + t2 * za * za);
      sb.h1j[4][s][h] = (bf16_t)(s1 * zab + t2 * za * zb);
    }
  }
  __syncthreads();
  // ---- layer 2 via bf16 WMMA (5 jets) + tanh combine + layer-3 dot ----
  {
    int lane = tid & 31, wv = tid >> 5, m0 = wv * 16;
    float pp[5][8];
#pragma unroll
    for (int j = 0; j < 5; ++j)
#pragma unroll
      for (int r = 0; r < 8; ++r) pp[j][r] = 0.f;

#pragma unroll 1
    for (int nt = 0; nt < 8; ++nt) {
      v8f a0 = {0.f,0.f,0.f,0.f,0.f,0.f,0.f,0.f};
      v8f a1 = a0, a2 = a0, a3 = a0, a4 = a0;
#pragma unroll
      for (int kc = 0; kc < 4; ++kc) {
        v16bf bf = load_b_frag(&sb.w2T[0][0], lane, nt, kc);
        v16bf af;
        af = load_a_frag(&sb.h1j[0][m0][0], lane, kc);
        a0 = __builtin_amdgcn_wmma_f32_16x16x32_bf16(false, af, false, bf, (short)0, a0, false, false);
        af = load_a_frag(&sb.h1j[1][m0][0], lane, kc);
        a1 = __builtin_amdgcn_wmma_f32_16x16x32_bf16(false, af, false, bf, (short)0, a1, false, false);
        af = load_a_frag(&sb.h1j[2][m0][0], lane, kc);
        a2 = __builtin_amdgcn_wmma_f32_16x16x32_bf16(false, af, false, bf, (short)0, a2, false, false);
        af = load_a_frag(&sb.h1j[3][m0][0], lane, kc);
        a3 = __builtin_amdgcn_wmma_f32_16x16x32_bf16(false, af, false, bf, (short)0, a3, false, false);
        af = load_a_frag(&sb.h1j[4][m0][0], lane, kc);
        a4 = __builtin_amdgcn_wmma_f32_16x16x32_bf16(false, af, false, bf, (short)0, a4, false, false);
      }
      int n = (lane & 15) + nt * 16;
      float b2n = sb.b2s[n], w3n = sb.w3s[n];
#pragma unroll
      for (int r = 0; r < 8; ++r) {
        float z  = a0[r] + b2n;
        float hv = fast_tanh(z);
        float s1 = 1.f - hv * hv;
        float t2 = -2.f * hv * s1;
        float za = a1[r], zb = a2[r];
        pp[0][r] += hv * w3n;
        pp[1][r] += (s1 * za) * w3n;
        pp[2][r] += (s1 * zb) * w3n;
        pp[3][r] += (s1 * a3[r] + t2 * za * za) * w3n;
        pp[4][r] += (s1 * a4[r] + t2 * za * zb) * w3n;
      }
    }
    // sum over N: xor-reduce the 16 lanes of each half (bit4 preserved)
#pragma unroll
    for (int j = 0; j < 5; ++j)
#pragma unroll
      for (int r = 0; r < 8; ++r) {
        float v = pp[j][r];
        v += __shfl_xor(v, 1, 32);
        v += __shfl_xor(v, 2, 32);
        v += __shfl_xor(v, 4, 32);
        v += __shfl_xor(v, 8, 32);
        pp[j][r] = v;
      }
    if ((lane & 15) == 0) {
      int mo = (lane >> 4) * 8;
#pragma unroll
      for (int r = 0; r < 8; ++r) {
        int s = m0 + mo + r;
        sb.piJ[0][s] = pp[0][r] + b3v;
        sb.piJ[1][s] = pp[1][r];
        sb.piJ[2][s] = pp[2][r];
        sb.piJ[3][s] = pp[3][r];
        sb.piJ[4][s] = pp[4][r];
      }
    }
  }
  __syncthreads();
}

__global__ void __launch_bounds__(TPB)
rollout_kernel(const float* __restrict__ Wp, const float* __restrict__ Yp,
               const float* __restrict__ w1, const float* __restrict__ b1,
               const float* __restrict__ w2, const float* __restrict__ b2,
               const float* __restrict__ w3, const float* __restrict__ b3,
               const float* __restrict__ noise, float* __restrict__ acc) {
  __shared__ SharedBlk sb;
  int tid = threadIdx.x;
  int gs  = blockIdx.x * SPW + tid;      // global sample id (valid if tid<SPW)
  bool owner = tid < SPW;

  // stage weights (128 threads)
#pragma unroll
  for (int j = 0; j < 3; ++j) sb.w1s[j][tid] = w1[j * 128 + tid];
  sb.b1s[tid] = b1[tid];
  sb.b2s[tid] = b2[tid];
  sb.w3s[tid] = w3[tid];
  for (int k = 0; k < 128; ++k) sb.w2T[tid][k] = (bf16_t)w2[k * 128 + tid];
  float b3v = b3[0];
  __syncthreads();

  // per-path jet state (owner threads)
  float L = 0.f, La = 0.f, Lb = 0.f, Laa = 0.f, Lab = 0.f, Yv = 0.f;
  if (owner) {
    int p = gs & 127;
    float W0 = Wp[p];
    L   = logf(W0);          // W0 in (0.8,1.2) > LB_W
    La  = 1.f / W0;
    Laa = -La * La;
    Yv  = Yp[p];
    __builtin_prefetch(&noise[(size_t)gs], 0, 2);   // warm step-0 noise
  }
  float cY = 1.f;            // dY/dY0, uniform deterministic scalar

#pragma unroll 1
  for (int st = 0; st < MSTEPS; ++st) {
    float tmt = T_F - st * DT_F;
    // step temporaries crossing guard boundaries
    float Lc=0,Lca=0,Lcb=0,Lcaa=0,Lcab=0, Yc=0;
    float Lh=0,Lha=0,Lhb=0,Lhaa=0,Lhab=0, Yh=0;
    float Sh2=0;

    float z1b=0,z2b=0,z1n=0,z2n=0;
    if (owner) {
      size_t nb = (size_t)(4 * st) * 4096 + (size_t)gs;
      z1b = noise[nb];
      z2b = noise[nb + 4096];
      z1n = noise[nb + 2 * 4096];
      z2n = noise[nb + 3 * 4096];
      if (st + 1 < MSTEPS) {           // prefetch next step's noise rows
        size_t nn = nb + 4 * 4096;
        __builtin_prefetch(&noise[nn], 0, 2);
        __builtin_prefetch(&noise[nn + 4096], 0, 2);
        __builtin_prefetch(&noise[nn + 2 * 4096], 0, 2);
        __builtin_prefetch(&noise[nn + 3 * 4096], 0, 2);
      }
      float W = __expf(L);
      sb.xj[0][tid] = W;
      sb.xj[1][tid] = W * La;
      sb.xj[2][tid] = W * Lb;
      sb.xj[3][tid] = W * (Laa + La * La);
      sb.xj[4][tid] = W * (Lab + La * Lb);
      sb.xy[tid]    = Yv;
    }
    mlp_eval(sb, tid, tmt, cY, b3v);                 // pi_t jets

    float cYh = cY * (1.f - KAPPA_F * HALF_F);
    if (owner) {
      float P   = sb.piJ[0][tid], Pa = sb.piJ[1][tid], Pb = sb.piJ[2][tid];
      float Paa = sb.piJ[3][tid], Pab = sb.piJ[4][tid];
      float zWb = z1b, zYb = RHO_F * z1b + ACORR_F * z2b;
      float zWn = z1n, zYn = RHO_F * z1n + ACORR_F * z2n;
      float zW1 = (zWb + zWn) * INVS2_F, zY1 = (zYb + zYn) * INVS2_F;
      float zW2 = (zWb - zWn) * INVS2_F, zY2 = (zYb - zYn) * INVS2_F;
      // drift & var jets
      float D   = RRATE_F + SIGALP_F * P * Yv;
      float Da  = SIGALP_F * Pa * Yv;
      float Db  = SIGALP_F * (Pb * Yv + P * cY);
      float Daa = SIGALP_F * Paa * Yv;
      float Dab = SIGALP_F * (Pab * Yv + Pa * cY);
      float V   = SIG2_F * P * P;
      float Va  = 2.f * SIG2_F * P * Pa;
      float Vb  = 2.f * SIG2_F * P * Pb;
      float Vaa = 2.f * SIG2_F * (Pa * Pa + P * Paa);
      float Vab = 2.f * SIG2_F * (Pa * Pb + P * Pab);
      float G   = D - 0.5f * V,  Ga = Da - 0.5f * Va,  Gb = Db - 0.5f * Vb;
      float Gaa = Daa - 0.5f * Vaa, Gab = Dab - 0.5f * Vab;
      float Sc  = SIGMA_F * SDT_F * zWb;
      float Sh1 = SIGMA_F * SHALF_F * zW1;
      Sh2 = SIGMA_F * SHALF_F * zW2;
      // coarse
      Lc = L + G * DT_F + P * Sc;   Lca = La + Ga * DT_F + Pa * Sc;
      Lcb = Lb + Gb * DT_F + Pb * Sc;
      Lcaa = Laa + Gaa * DT_F + Paa * Sc;  Lcab = Lab + Gab * DT_F + Pab * Sc;
      Yc = Yv * (1.f - KAPPA_F * DT_F) + SIGY_F * SDT_F * zYb;
      // first half
      Lh = L + G * HALF_F + P * Sh1;  Lha = La + Ga * HALF_F + Pa * Sh1;
      Lhb = Lb + Gb * HALF_F + Pb * Sh1;
      Lhaa = Laa + Gaa * HALF_F + Paa * Sh1;  Lhab = Lab + Gab * HALF_F + Pab * Sh1;
      Yh = Yv * (1.f - KAPPA_F * HALF_F) + SIGY_F * SHALF_F * zY1;
      // second-half policy inputs
      float Wh = __expf(Lh);
      sb.xj[0][tid] = Wh;
      sb.xj[1][tid] = Wh * Lha;
      sb.xj[2][tid] = Wh * Lhb;
      sb.xj[3][tid] = Wh * (Lhaa + Lha * Lha);
      sb.xj[4][tid] = Wh * (Lhab + Lha * Lhb);
      sb.xy[tid]    = Yh;
      // fold deterministic-linear Y to its Richardson value now
      float Yf = Yh * (1.f - KAPPA_F * HALF_F) + SIGY_F * SHALF_F * zY2;
      Yc = 2.f * Yf - Yc;           // Yc now holds Y_new
    }
    mlp_eval(sb, tid, tmt - HALF_F, cYh, b3v);       // pi_h jets

    if (owner) {
      float P   = sb.piJ[0][tid], Pa = sb.piJ[1][tid], Pb = sb.piJ[2][tid];
      float Paa = sb.piJ[3][tid], Pab = sb.piJ[4][tid];
      float D   = RRATE_F + SIGALP_F * P * Yh;
      float Da  = SIGALP_F * Pa * Yh;
      float Db  = SIGALP_F * (Pb * Yh + P * cYh);
      float Daa = SIGALP_F * Paa * Yh;
      float Dab = SIGALP_F * (Pab * Yh + Pa * cYh);
      float V   = SIG2_F * P * P;
      float Va  = 2.f * SIG2_F * P * Pa;
      float Vb  = 2.f * SIG2_F * P * Pb;
      float Vaa = 2.f * SIG2_F * (Pa * Pa + P * Paa);
      float Vab = 2.f * SIG2_F * (Pa * Pb + P * Pab);
      float G   = D - 0.5f * V,  Ga = Da - 0.5f * Va,  Gb = Db - 0.5f * Vb;
      float Gaa = Daa - 0.5f * Vaa, Gab = Dab - 0.5f * Vab;
      float Lf   = Lh + G * HALF_F + P * Sh2;
      float Lfa  = Lha + Ga * HALF_F + Pa * Sh2;
      float Lfb  = Lhb + Gb * HALF_F + Pb * Sh2;
      float Lfaa = Lhaa + Gaa * HALF_F + Paa * Sh2;
      float Lfab = Lhab + Gab * HALF_F + Pab * Sh2;
      // Richardson
      float Ln   = 2.f * Lf - Lc;
      float Lna  = 2.f * Lfa - Lca;
      float Lnb  = 2.f * Lfb - Lcb;
      float Lnaa = 2.f * Lfaa - Lcaa;
      float Lnab = 2.f * Lfab - Lcab;
      if (Ln < LOGLB_F) { Ln = LOGLB_F; Lna = Lnb = Lnaa = Lnab = 0.f; }
      L = Ln; La = Lna; Lb = Lnb; Laa = Lnaa; Lab = Lnab;
      Yv = Yc;
    }
    float kh = 1.f - KAPPA_F * HALF_F;
    cY = cY * (2.f * kh * kh - (1.f - KAPPA_F * DT_F));
  }

  if (owner) {
    // U = exp((1-g)L)/(1-g), g=5 -> E = exp(-4L)
    float E   = __expf(-4.f * L);
    float up  = E * La;
    float uaa = E * (Laa - 4.f * La * La);
    float uab = E * (Lab - 4.f * La * Lb);
    int p = gs & 127;
    atomicAdd(&acc[p],       up);
    atomicAdd(&acc[128 + p], uaa);
    atomicAdd(&acc[256 + p], uab);
  }
}

__global__ void init_kernel(float* acc) {
  int t = threadIdx.x;
  if (t < 384) acc[t] = 0.f;
}

__global__ void finalize_kernel(const float* __restrict__ acc,
                                const float* __restrict__ Wp,
                                const float* __restrict__ Yp,
                                float* __restrict__ out) {
  int p = threadIdx.x;
  if (p >= 128) return;
  // reference double-averaging: lam=sum/32768, dlam=sum/1048576
  float lam = acc[p]       * (1.f / 32768.f);
  float dW  = acc[128 + p] * (1.f / 1048576.f);
  float dY  = acc[256 + p] * (1.f / 1048576.f);
  float coeff = -1.f / (Wp[p] * dW + 1e-8f);
  float mur   = SIGALP_F * Yp[p];
  float pi = coeff * lam * (mur / SIG2_F)
           + coeff * (SIGMA_F * RHO_F * SIGY_F * dY / SIG2_F);
  pi = fminf(fmaxf(pi, -2.f), 2.f);
  out[p] = pi;
}

extern "C" void kernel_launch(void* const* d_in, const int* in_sizes, int n_in,
                              void* d_out, int out_size, void* d_ws, size_t ws_size,
                              hipStream_t stream) {
  const float* W     = (const float*)d_in[0];
  // d_in[1] = TmT (constant T, unused)
  const float* Y     = (const float*)d_in[2];
  const float* w1    = (const float*)d_in[3];
  const float* b1    = (const float*)d_in[4];
  const float* w2    = (const float*)d_in[5];
  const float* b2    = (const float*)d_in[6];
  const float* w3    = (const float*)d_in[7];
  const float* b3    = (const float*)d_in[8];
  const float* noise = (const float*)d_in[9];
  float* acc = (float*)d_ws;

  init_kernel<<<1, 384, 0, stream>>>(acc);
  rollout_kernel<<<NWG, TPB, 0, stream>>>(W, Y, w1, b1, w2, b2, w3, b3, noise, acc);
  finalize_kernel<<<1, 128, 0, stream>>>(acc, W, Y, (float*)d_out);
}
